// PRISMTrunk_58634893525830
// MI455X (gfx1250) — matrix operand order repaired and checked by reading, not compiled, not run on hardware
//
#include <hip/hip_runtime.h>
#include <math.h>

// ---------------------------------------------------------------------------
// PRISM trunk for MI455X (gfx1250, wave32). All GEMMs via V_WMMA_F32_16X16X32_F16
// (f16 operands, f32 accumulate). Weights converted once per launch to f16 in
// workspace, transposed so B operand is [N][K] row-major. TCN conv uses
// LDS-staged activations with zero-padded boundary rows so the im2col K-loop
// is branch-free (pure ds_load + v_wmma).
// ---------------------------------------------------------------------------

typedef __attribute__((ext_vector_type(16))) _Float16 v16h;
typedef __attribute__((ext_vector_type(8)))  float    v8f;

__device__ __forceinline__ float gelu_f(float x) {
  return 0.5f * x * (1.0f + erff(x * 0.7071067811865476f));
}

// A-tile loader: A is [M][K] row-major f16. ISA 16-bit A 16x32 layout:
// lanes 0-15: row M=lane, K = {0..7, 16..23}; lanes 16-31: row M=lane-16, K = {8..15, 24..31}.
__device__ __forceinline__ v16h load_a16(const _Float16* A, int lda, int row0, int k0) {
  int lane = threadIdx.x & 31;
  int r = lane & 15, hi = lane >> 4;
  const _Float16* p = A + (size_t)(row0 + r) * lda + (k0 + hi * 8);
  union { v16h v; float4 q[2]; } u;
  u.q[0] = *(const float4*)(p);
  u.q[1] = *(const float4*)(p + 16);
  return u.v;
}

// B-tile loader: B source stored [N][K] row-major f16 (i.e. math-B transposed).
// lanes 0-15: col N=lane, K = {0..7, 16..23}; lanes 16-31: col N=lane-16, K = {8..15, 24..31}.
__device__ __forceinline__ v16h load_b16(const _Float16* B, int ldb, int col0, int k0) {
  int lane = threadIdx.x & 31;
  int c = lane & 15, hi = lane >> 4;
  const _Float16* p = B + (size_t)(col0 + c) * ldb + (k0 + hi * 8);
  union { v16h v; float4 q[2]; } u;
  u.q[0] = *(const float4*)(p);
  u.q[1] = *(const float4*)(p + 16);
  return u.v;
}

__device__ __forceinline__ v8f wmma16(v16h a, v16h b, v8f c) {
  return __builtin_amdgcn_wmma_f32_16x16x32_f16(false, a, false, b, (short)0, c, false, false);
}

#define V8F_ZERO {0.f,0.f,0.f,0.f,0.f,0.f,0.f,0.f}

// ---------------------------------------------------------------------------
// Prep kernels
// ---------------------------------------------------------------------------

// feat (256,64,64) -> feat_t (64*64, 256) so channel gathers are coalesced.
__global__ void k_feat_t(const float* __restrict__ feat, float* __restrict__ ft) {
  int i = blockIdx.x * 256 + threadIdx.x;
  if (i >= 64 * 64 * 256) return;
  int c = i & 255; int yx = i >> 8;
  ft[i] = feat[(size_t)c * 4096 + yx];
}

// dst[r][c] (c<ldd) = (c<C) ? src[c*R + r] : 0   (f32->f16 transpose+pad)
__global__ void k_convT(_Float16* __restrict__ dst, const float* __restrict__ src,
                        int R, int C, int ldd) {
  int i = blockIdx.x * 256 + threadIdx.x;
  if (i >= R * ldd) return;
  int r = i / ldd, c = i - r * ldd;
  dst[i] = (c < C) ? (_Float16)src[(size_t)c * R + r] : (_Float16)0.f;
}

__global__ void k_convP(_Float16* __restrict__ dst, const float* __restrict__ src, int n) {
  int i = blockIdx.x * 256 + threadIdx.x;
  if (i < n) dst[i] = (_Float16)src[i];
}

// c1_w (768,768,3) OIH -> Wcol[co][k*768+ci] f16 (im2col ordering)
__global__ void k_convC1(_Float16* __restrict__ dst, const float* __restrict__ src) {
  int i = blockIdx.x * 256 + threadIdx.x;
  if (i >= 768 * 2304) return;
  int co = i / 2304; int rem = i - co * 2304;
  int k = rem / 768; int ci = rem - k * 768;
  dst[i] = (_Float16)src[(size_t)co * 2304 + ci * 3 + k];
}

// Positional embedding pe[32][32] (shared by all boxes).
__global__ void k_pe(const float* w1, const float* b1, const float* w2, const float* b2,
                     float* __restrict__ pe) {
  for (int t = threadIdx.x; t < 1024; t += 256) {
    int p = t >> 5, d = t & 31;
    float pos = fabsf((float)p * (1.f / 31.f) - 0.5f) * 2.f;
    float acc = b2[d];
    for (int dd = 0; dd < 32; ++dd) {
      float h = gelu_f(pos * w1[dd] + b1[dd]);
      acc += h * w2[dd * 32 + d];
    }
    pe[t] = acc;
  }
}

// Sampling: block = (m,p); thread = channel. Writes Y[m][p][j*256+c] (f32).
__global__ void k_sample(const float* __restrict__ ft, const float* __restrict__ boxes,
                         float* __restrict__ Y) {
  int mp = blockIdx.x; int m = mp >> 5, p = mp & 31;
  int c = threadIdx.x;
  float cx = boxes[m*4+0] * 1024.f, cy = boxes[m*4+1] * 1024.f;
  float bw = fmaxf(boxes[m*4+2] * 1024.f, 1.f);
  float bh = fmaxf(boxes[m*4+3] * 1024.f, 1.f);
  float L = sqrtf(bw*bw + bh*bh);
  float inv = 1.f / (L + 1e-6f);
  float ux = bw*inv, uy = bh*inv;
  float nx = -uy, ny = ux;
  float lin = -0.5f + (float)p * (1.f/31.f);
  float tv = (lin + 0.015625f) * L - 0.5f * L;
  float sh = fminf(bw, bh);
  float delta = 0.35f * sh, aa = 0.25f * sh;
  size_t yb = ((size_t)m*32 + p) * 768;
  for (int j = 0; j < 3; ++j) {
    float acc = 0.f;
    float db = (j == 1) ? -delta : ((j == 2) ? delta : 0.f);
    for (int k = 0; k < 8; ++k) {
      float lam = (-1.f + (float)k * (2.f/7.f)) * aa;
      float off = lam + db;
      float gx = fminf(fmaxf(cx + tv*ux + off*nx, 0.f), 1023.f);
      float gy = fminf(fmaxf(cy + tv*uy + off*ny, 0.f), 1023.f);
      float xf = gx * (63.f/1023.f), yf = gy * (63.f/1023.f);
      float x0 = floorf(xf), y0 = floorf(yf);
      float wx = xf - x0, wy = yf - y0;
      int ix0 = min(max((int)x0, 0), 63); int ix1 = min(ix0 + 1, 63);
      int iy0 = min(max((int)y0, 0), 63); int iy1 = min(iy0 + 1, 63);
      float v00 = ft[((size_t)iy0*64 + ix0)*256 + c];
      float v01 = ft[((size_t)iy0*64 + ix1)*256 + c];
      float v10 = ft[((size_t)iy1*64 + ix0)*256 + c];
      float v11 = ft[((size_t)iy1*64 + ix1)*256 + c];
      acc += (1.f-wy)*((1.f-wx)*v00 + wx*v01) + wy*((1.f-wx)*v10 + wx*v11);
    }
    Y[yb + j*256 + c] = acc * 0.125f;
  }
}

// ---------------------------------------------------------------------------
// LayerNorm kernels
// ---------------------------------------------------------------------------

__global__ void k_ln768(const float* __restrict__ Y, const float* g, const float* b,
                        _Float16* __restrict__ H) {
  __shared__ float red[256];
  int row = blockIdx.x; int tid = threadIdx.x;
  const float* x = Y + (size_t)row * 768;
  float x0 = x[tid], x1 = x[tid+256], x2 = x[tid+512];
  red[tid] = x0 + x1 + x2; __syncthreads();
  for (int s = 128; s > 0; s >>= 1) { if (tid < s) red[tid] += red[tid+s]; __syncthreads(); }
  float mean = red[0] * (1.f/768.f); __syncthreads();
  float d0 = x0-mean, d1 = x1-mean, d2 = x2-mean;
  red[tid] = d0*d0 + d1*d1 + d2*d2; __syncthreads();
  for (int s = 128; s > 0; s >>= 1) { if (tid < s) red[tid] += red[tid+s]; __syncthreads(); }
  float rstd = rsqrtf(red[0] * (1.f/768.f) + 1e-5f);
  _Float16* h = H + (size_t)row * 768;
  h[tid]     = (_Float16)(d0*rstd*g[tid]     + b[tid]);
  h[tid+256] = (_Float16)(d1*rstd*g[tid+256] + b[tid+256]);
  h[tid+512] = (_Float16)(d2*rstd*g[tid+512] + b[tid+512]);
}

__global__ void k_ln256(const float* __restrict__ Y, const float* g, const float* b,
                        _Float16* __restrict__ dst, int dstride, float* __restrict__ dst32) {
  __shared__ float red[256];
  int row = blockIdx.x; int tid = threadIdx.x;
  float x = Y[(size_t)row * 768 + tid];
  red[tid] = x; __syncthreads();
  for (int s = 128; s > 0; s >>= 1) { if (tid < s) red[tid] += red[tid+s]; __syncthreads(); }
  float mean = red[0] * (1.f/256.f); __syncthreads();
  float d = x - mean;
  red[tid] = d*d; __syncthreads();
  for (int s = 128; s > 0; s >>= 1) { if (tid < s) red[tid] += red[tid+s]; __syncthreads(); }
  float rstd = rsqrtf(red[0] * (1.f/256.f) + 1e-5f);
  float v = d*rstd*g[tid] + b[tid];
  dst[(size_t)row * dstride + tid] = (_Float16)v;
  if (dst32) dst32[(size_t)row * 256 + tid] = v;
}

// ---------------------------------------------------------------------------
// TCN GEMMs: out[co][t] = sum_kk Wcol[co][kk] * B(kk,t)
// Block = (box, co-half). Activations staged in LDS with 2 zero-pad rows on
// each side so the dilation-shifted B loads need no bounds check.
// ---------------------------------------------------------------------------

__global__ void __launch_bounds__(256) k_tcn1(const _Float16* __restrict__ Wc1,
                                              const float* __restrict__ b1,
                                              const _Float16* __restrict__ H,
                                              _Float16* __restrict__ H1, int dil) {
  __shared__ __align__(16) _Float16 Hs[36 * 768];  // rows 0-1 and 34-35 are zero
  int m = blockIdx.x >> 1;
  int half = blockIdx.x & 1;
  const _Float16* Hm = H + (size_t)m * 32 * 768;
  // zero-pad rows: 4 rows * 768 halfs = 384 x float4
  for (int i = threadIdx.x; i < 384; i += 256) {
    int r = i / 96; int cc = (i - r * 96) * 8;
    int row = (r < 2) ? r : (32 + r);
    *(float4*)(Hs + (size_t)row * 768 + cc) = make_float4(0.f, 0.f, 0.f, 0.f);
  }
  // data rows 2..33: 32 rows * 768 halfs = 3072 x float4
  for (int i = threadIdx.x; i < 3072; i += 256) {
    int row = i / 96; int cc = (i - row * 96) * 8;
    *(float4*)(Hs + (size_t)(row + 2) * 768 + cc) = *(const float4*)(Hm + (size_t)row * 768 + cc);
  }
  __syncthreads();

  int wave = threadIdx.x >> 5;
  int lane = threadIdx.x & 31; int cN = lane & 15; int hi = lane >> 4;
  // warm L2 with this wave's weight rows
  __builtin_prefetch(Wc1 + (size_t)(half * 384 + wave * 48 + cN) * 2304, 0, 1);
  for (int i = 0; i < 6; ++i) {
    int tile = wave * 6 + i;                 // 48 tiles in this co-half
    int mt = half * 24 + (tile >> 1); int nt = tile & 1;
    int m0 = mt * 16, n0 = nt * 16;
    v8f acc = V8F_ZERO;
    const _Float16* arow0 = Wc1 + (size_t)(m0 + cN) * 2304 + hi * 8;
    for (int tap = 0; tap < 3; ++tap) {
      const _Float16* arow = arow0 + tap * 768;
      const _Float16* brow = Hs + (size_t)(n0 + cN + (tap - 1) * dil + 2) * 768 + hi * 8;
      for (int kb = 0; kb < 24; ++kb) {
        union { v16h v; float4 q[2]; } ua, ub;
        const _Float16* pa = arow + kb * 32;
        ua.q[0] = *(const float4*)(pa);
        ua.q[1] = *(const float4*)(pa + 16);
        const _Float16* pb = brow + kb * 32;
        ub.q[0] = *(const float4*)(pb);
        ub.q[1] = *(const float4*)(pb + 16);
        acc = wmma16(ua.v, ub.v, acc);
      }
    }
    int t = n0 + cN; int co0 = m0 + hi * 8;
    union { _Float16 h[8]; float4 f4; } u;
    for (int rr = 0; rr < 8; ++rr) u.h[rr] = (_Float16)gelu_f(acc[rr] + b1[co0 + rr]);
    *(float4*)(H1 + ((size_t)m * 32 + t) * 768 + co0) = u.f4;
  }
}

__global__ void __launch_bounds__(256) k_tcn2(const _Float16* __restrict__ Wc2,
                                              const float* __restrict__ b2,
                                              const _Float16* __restrict__ H1,
                                              float* __restrict__ Y) {
  __shared__ __align__(16) _Float16 Hs[32 * 768];
  int m = blockIdx.x >> 1;
  int half = blockIdx.x & 1;
  const _Float16* Hm = H1 + (size_t)m * 32 * 768;
  for (int i = threadIdx.x; i < 3072; i += 256) {
    int row = i / 96; int cc = (i - row * 96) * 8;
    *(float4*)(Hs + (size_t)row * 768 + cc) = *(const float4*)(Hm + (size_t)row * 768 + cc);
  }
  __syncthreads();

  int wave = threadIdx.x >> 5;
  int lane = threadIdx.x & 31; int cN = lane & 15; int hi = lane >> 4;
  __builtin_prefetch(Wc2 + (size_t)(half * 384 + wave * 48 + cN) * 768, 0, 1);
  for (int i = 0; i < 6; ++i) {
    int tile = wave * 6 + i;
    int mt = half * 24 + (tile >> 1); int nt = tile & 1;
    int m0 = mt * 16, n0 = nt * 16;
    v8f acc = V8F_ZERO;
    for (int kb = 0; kb < 24; ++kb)
      acc = wmma16(load_a16(Wc2, 768, m0, kb * 32), load_b16(Hs, 768, n0, kb * 32), acc);
    int t = n0 + cN; int co0 = m0 + hi * 8;
    float* y = Y + ((size_t)m * 32 + t) * 768 + co0;
    for (int rr = 0; rr < 8; ++rr) y[rr] += acc[rr] + b2[co0 + rr];
  }
}

// f16 mirrors of Y (center+side rows) for attention GEMM operands.
__global__ void k_y2h(const float* __restrict__ Y, _Float16* __restrict__ Yh,
                      _Float16* __restrict__ Sh) {
  int i = blockIdx.x * 256 + threadIdx.x;
  if (i < 3145728) {
    Yh[i] = (_Float16)Y[i];
  } else if (i < 3145728 + 2097152) {
    int t = i - 3145728;
    int c = t & 255; int pp = (t >> 8) & 63; int m = t >> 14;
    Sh[t] = (_Float16)Y[((size_t)m * 32 + (pp & 31)) * 768 + 256 + (pp >> 5) * 256 + c];
  }
}

// ---------------------------------------------------------------------------
// Attention: q/k/vT projections, per-head softmax attention, o-proj.
// ---------------------------------------------------------------------------

__global__ void k_qkv(const _Float16* __restrict__ Yh, const _Float16* __restrict__ Sh,
                      const _Float16* __restrict__ WqT, const float* __restrict__ qb,
                      const _Float16* __restrict__ WkT, const float* __restrict__ kbias,
                      const _Float16* __restrict__ WvT, const float* __restrict__ vb,
                      _Float16* __restrict__ Qw, _Float16* __restrict__ Kw,
                      _Float16* __restrict__ VT) {
  int wave = threadIdx.x >> 5;
  int gt = blockIdx.x * 8 + wave;
  if (gt >= 20480) return;
  int lane = threadIdx.x & 31; int cN = lane & 15; int hi = lane >> 4;
  v8f acc = V8F_ZERO;
  if (gt < 4096) {                    // q = center @ Wq : [32][256]
    int m = gt >> 5; int r = gt & 31; int mt = r >> 4; int nt = r & 15;
    const _Float16* A = Yh + (size_t)m * 32 * 768;
    for (int kb = 0; kb < 8; ++kb)
      acc = wmma16(load_a16(A, 768, mt*16, kb*32), load_b16(WqT, 256, nt*16, kb*32), acc);
    int p0 = mt*16 + hi*8; int co = nt*16 + cN;
    for (int rr = 0; rr < 8; ++rr)
      Qw[((size_t)m*32 + p0 + rr)*256 + co] = (_Float16)(acc[rr] + qb[co]);
  } else if (gt < 12288) {            // k = side @ Wk : [64][256]
    int g = gt - 4096; int m = g >> 6; int r = g & 63; int mt = r >> 4; int nt = r & 15;
    const _Float16* A = Sh + (size_t)m * 64 * 256;
    for (int kb = 0; kb < 8; ++kb)
      acc = wmma16(load_a16(A, 256, mt*16, kb*32), load_b16(WkT, 256, nt*16, kb*32), acc);
    int p0 = mt*16 + hi*8; int co = nt*16 + cN;
    for (int rr = 0; rr < 8; ++rr)
      Kw[((size_t)m*64 + p0 + rr)*256 + co] = (_Float16)(acc[rr] + kbias[co]);
  } else {                            // vT[d][pk] = Wv^T @ side^T : [256][64]
    int g = gt - 12288; int m = g >> 6; int r = g & 63; int mt = r >> 2; int nt = r & 3;
    const _Float16* B = Sh + (size_t)m * 64 * 256;
    for (int kb = 0; kb < 8; ++kb)
      acc = wmma16(load_a16(WvT, 256, mt*16, kb*32), load_b16(B, 256, nt*16, kb*32), acc);
    int d0 = mt*16 + hi*8; int pk = nt*16 + cN;
    for (int rr = 0; rr < 8; ++rr)
      VT[(size_t)m*16384 + (size_t)(d0 + rr)*64 + pk] = (_Float16)(acc[rr] + vb[d0 + rr]);
  }
}

__global__ void __launch_bounds__(64) k_attn(const _Float16* __restrict__ Qw,
                                             const _Float16* __restrict__ Kw,
                                             const _Float16* __restrict__ VT,
                                             _Float16* __restrict__ Ow) {
  __shared__ __align__(16) float    sc[32 * 64];
  __shared__ __align__(16) _Float16 att[32 * 64];
  int m = blockIdx.x >> 3; int h = blockIdx.x & 7;
  int wave = threadIdx.x >> 5; int lane = threadIdx.x & 31;
  int cN = lane & 15; int hi = lane >> 4;
  const _Float16* Qm = Qw + (size_t)m * 32 * 256;
  const _Float16* Km = Kw + (size_t)m * 64 * 256;
  // scores: 2 p-tiles x 4 pk-tiles, K = dh = 32 (one block)
  for (int tt = wave * 4; tt < wave * 4 + 4; ++tt) {
    int mt = tt >> 2; int nt = tt & 3;
    v8f acc = V8F_ZERO;
    acc = wmma16(load_a16(Qm, 256, mt*16, h*32), load_b16(Km, 256, nt*16, h*32), acc);
    int p0 = mt*16 + hi*8; int pk = nt*16 + cN;
    for (int rr = 0; rr < 8; ++rr)
      sc[(p0 + rr)*64 + pk] = acc[rr] * 0.17677669529663687f;  // 1/sqrt(32)
  }
  __syncthreads();
  if (threadIdx.x < 32) {
    int p = threadIdx.x;
    float mx = -1e30f;
    for (int j = 0; j < 64; ++j) mx = fmaxf(mx, sc[p*64 + j]);
    float s = 0.f;
    for (int j = 0; j < 64; ++j) { float e = __expf(sc[p*64 + j] - mx); sc[p*64 + j] = e; s += e; }
    float inv = 1.f / s;
    for (int j = 0; j < 64; ++j) att[p*64 + j] = (_Float16)(sc[p*64 + j] * inv);
  }
  __syncthreads();
  // o = att @ v : A = att [32][64], B = vT rows h*32..h*32+31
  const _Float16* Vh = VT + (size_t)m * 16384 + (size_t)h * 32 * 64;
  for (int tt = wave * 2; tt < wave * 2 + 2; ++tt) {
    int mt = tt >> 1; int nt = tt & 1;
    v8f acc = V8F_ZERO;
    for (int kb = 0; kb < 2; ++kb)
      acc = wmma16(load_a16((const _Float16*)att, 64, mt*16, kb*32),
                   load_b16(Vh, 64, nt*16, kb*32), acc);
    int p0 = mt*16 + hi*8; int d = nt*16 + cN;
    for (int rr = 0; rr < 8; ++rr)
      Ow[((size_t)m*32 + p0 + rr)*256 + h*32 + d] = (_Float16)acc[rr];
  }
}

// Generic [32 rows]x[N] GEMM kernels (A f16 global/ws, B = transposed weights)
__global__ void k_oproj(const _Float16* __restrict__ Ow, const _Float16* __restrict__ WoT,
                        const float* __restrict__ ob, float* __restrict__ Y) {
  int wave = threadIdx.x >> 5; int gt = blockIdx.x * 8 + wave;
  if (gt >= 4096) return;
  int m = gt >> 5; int r = gt & 31; int mt = r >> 4; int nt = r & 15;
  const _Float16* A = Ow + (size_t)m * 32 * 256;
  v8f acc = V8F_ZERO;
  for (int kb = 0; kb < 8; ++kb)
    acc = wmma16(load_a16(A, 256, mt*16, kb*32), load_b16(WoT, 256, nt*16, kb*32), acc);
  int lane = threadIdx.x & 31; int cN = lane & 15; int hi = lane >> 4;
  int p0 = mt*16 + hi*8; int co = nt*16 + cN;
  for (int rr = 0; rr < 8; ++rr)
    Y[((size_t)m*32 + p0 + rr)*768 + co] += acc[rr] + ob[co];
}

__global__ void k_ffn1(const _Float16* __restrict__ HL, const _Float16* __restrict__ W1T,
                       const float* __restrict__ b1, _Float16* __restrict__ T1) {
  int wave = threadIdx.x >> 5; int gt = blockIdx.x * 8 + wave;
  if (gt >= 16384) return;
  int m = gt >> 7; int r = gt & 127; int mt = r >> 6; int nt = r & 63;
  const _Float16* A = HL + (size_t)m * 32 * 256;
  v8f acc = V8F_ZERO;
  for (int kb = 0; kb < 8; ++kb)
    acc = wmma16(load_a16(A, 256, mt*16, kb*32), load_b16(W1T, 256, nt*16, kb*32), acc);
  int lane = threadIdx.x & 31; int cN = lane & 15; int hi = lane >> 4;
  int p0 = mt*16 + hi*8; int co = nt*16 + cN;
  for (int rr = 0; rr < 8; ++rr)
    T1[((size_t)m*32 + p0 + rr)*1024 + co] = (_Float16)gelu_f(acc[rr] + b1[co]);
}

__global__ void k_ffn2(const _Float16* __restrict__ T1, const _Float16* __restrict__ W2T,
                       const float* __restrict__ b2, float* __restrict__ Y) {
  int wave = threadIdx.x >> 5; int gt = blockIdx.x * 8 + wave;
  if (gt >= 4096) return;
  int m = gt >> 5; int r = gt & 31; int mt = r >> 4; int nt = r & 15;
  const _Float16* A = T1 + (size_t)m * 32 * 1024;
  v8f acc = V8F_ZERO;
  for (int kb = 0; kb < 32; ++kb)
    acc = wmma16(load_a16(A, 1024, mt*16, kb*32), load_b16(W2T, 1024, nt*16, kb*32), acc);
  int lane = threadIdx.x & 31; int cN = lane & 15; int hi = lane >> 4;
  int p0 = mt*16 + hi*8; int co = nt*16 + cN;
  for (int rr = 0; rr < 8; ++rr)
    Y[((size_t)m*32 + p0 + rr)*768 + co] += acc[rr] + b2[co];
}

// ---------------------------------------------------------------------------
// SRM branch + modulation
// ---------------------------------------------------------------------------

__global__ void k_srm_in(const float* __restrict__ Y, _Float16* __restrict__ SIN) {
  int row = blockIdx.x; int c = threadIdx.x;
  float l = Y[(size_t)row*768 + 256 + c];
  float r = Y[(size_t)row*768 + 512 + c];
  _Float16* s = SIN + (size_t)row * 1024;
  s[c] = (_Float16)l; s[256 + c] = (_Float16)r;
  s[512 + c] = (_Float16)(l - r); s[768 + c] = (_Float16)(l + r);
}

__global__ void k_srm1(const _Float16* __restrict__ SIN, const _Float16* __restrict__ W1T,
                       const float* __restrict__ b1, _Float16* __restrict__ S1) {
  int wave = threadIdx.x >> 5; int gt = blockIdx.x * 8 + wave;
  if (gt >= 2048) return;
  int m = gt >> 4; int r = gt & 15; int mt = r >> 3; int nt = r & 7;
  const _Float16* A = SIN + (size_t)m * 32 * 1024;
  v8f acc = V8F_ZERO;
  for (int kb = 0; kb < 32; ++kb)
    acc = wmma16(load_a16(A, 1024, mt*16, kb*32), load_b16(W1T, 1024, nt*16, kb*32), acc);
  int lane = threadIdx.x & 31; int cN = lane & 15; int hi = lane >> 4;
  int p0 = mt*16 + hi*8; int co = nt*16 + cN;
  for (int rr = 0; rr < 8; ++rr)
    S1[((size_t)m*32 + p0 + rr)*128 + co] = (_Float16)gelu_f(acc[rr] + b1[co]);
}

__global__ void k_srm2(const _Float16* __restrict__ S1, const _Float16* __restrict__ W2T,
                       const float* __restrict__ b2, _Float16* __restrict__ MODIN) {
  int wave = threadIdx.x >> 5; int gt = blockIdx.x * 8 + wave;
  if (gt >= 2048) return;
  int m = gt >> 4; int r = gt & 15; int mt = r >> 3; int nt = r & 7;
  const _Float16* A = S1 + (size_t)m * 32 * 128;
  v8f acc = V8F_ZERO;
  for (int kb = 0; kb < 4; ++kb)
    acc = wmma16(load_a16(A, 128, mt*16, kb*32), load_b16(W2T, 128, nt*16, kb*32), acc);
  int lane = threadIdx.x & 31; int cN = lane & 15; int hi = lane >> 4;
  int p0 = mt*16 + hi*8; int co = nt*16 + cN;
  for (int rr = 0; rr < 8; ++rr)
    MODIN[((size_t)m*32 + p0 + rr)*448 + 256 + co] = (_Float16)(acc[rr] + b2[co]);
}

__global__ void __launch_bounds__(64) k_modin_pe(const float* __restrict__ pe,
                                                 _Float16* __restrict__ MODIN) {
  int row = blockIdx.x; int p = row & 31; int t = threadIdx.x;
  if (t < 32) MODIN[(size_t)row*448 + 384 + t] = (_Float16)pe[p*32 + t];
  else        MODIN[(size_t)row*448 + 416 + (t - 32)] = (_Float16)0.f;   // K padding
}

__global__ void k_mod(const _Float16* __restrict__ MODIN,
                      const _Float16* __restrict__ WcT, const float* __restrict__ cb,
                      const _Float16* __restrict__ WeT, const float* __restrict__ eb,
                      float* __restrict__ MODC, float* __restrict__ MODE) {
  int wave = threadIdx.x >> 5; int gt = blockIdx.x * 8 + wave;
  if (gt >= 16384) return;
  const _Float16* B; const float* bias; float* OUT;
  int g = gt;
  if (g < 8192) { B = WcT; bias = cb; OUT = MODC; }
  else          { g -= 8192; B = WeT; bias = eb; OUT = MODE; }
  int m = g >> 6; int r = g & 63; int mt = r >> 5; int nt = r & 31;
  const _Float16* A = MODIN + (size_t)m * 32 * 448;
  v8f acc = V8F_ZERO;
  for (int kb = 0; kb < 14; ++kb)
    acc = wmma16(load_a16(A, 448, mt*16, kb*32), load_b16(B, 448, nt*16, kb*32), acc);
  int lane = threadIdx.x & 31; int cN = lane & 15; int hi = lane >> 4;
  int p0 = mt*16 + hi*8; int co = nt*16 + cN;
  for (int rr = 0; rr < 8; ++rr)
    OUT[((size_t)m*32 + p0 + rr)*512 + co] = acc[rr] + bias[co];
}

__global__ void k_combine(const _Float16* __restrict__ MODIN, const float* __restrict__ gw,
                          const float* __restrict__ gb, const float* __restrict__ CLN,
                          const float* __restrict__ MODC, const float* __restrict__ MODE,
                          float* __restrict__ Y) {
  __shared__ float red[256];
  int row = blockIdx.x; int c = threadIdx.x;
  const _Float16* mi = MODIN + (size_t)row * 448;
  float part = (float)mi[c] * gw[c];
  if (c < 160) part += (float)mi[256 + c] * gw[256 + c];
  red[c] = part; __syncthreads();
  for (int s = 128; s > 0; s >>= 1) { if (c < s) red[c] += red[c+s]; __syncthreads(); }
  float alpha = 1.f / (1.f + __expf(-(red[0] + gb[0])));
  float cl = CLN[(size_t)row*256 + c];
  float gc = MODC[(size_t)row*512 + c],       bc = MODC[(size_t)row*512 + 256 + c];
  float ge = MODE[(size_t)row*512 + c],       be = MODE[(size_t)row*512 + 256 + c];
  float vc = (1.f + gc)*cl + bc;
  float ve = (1.f + ge)*cl + be;
  Y[(size_t)row*768 + c] = (1.f - alpha)*vc + alpha*ve;
}

// ---------------------------------------------------------------------------
// Outputs
// ---------------------------------------------------------------------------

__global__ void k_out_copy(const float* __restrict__ Y, float* __restrict__ out) {
  int i = blockIdx.x * 256 + threadIdx.x;
  if (i < 3145728) out[i] = Y[i];
}

__global__ void k_pool(const float* __restrict__ Y, const float* __restrict__ cw,
                       const float* __restrict__ cbias, float* __restrict__ out) {
  __shared__ float pl[256];
  int m = blockIdx.x; int c = threadIdx.x;
  float s = 0.f;
  for (int p = 12; p < 21; ++p)
    for (int j = 0; j < 3; ++j)
      s += Y[((size_t)m*32 + p)*768 + j*256 + c];
  float pooled = s * (1.f / 27.f);
  out[3145728 + m*256 + c] = pooled;
  pl[c] = pooled; __syncthreads();
  if (c < 2) {
    float a = cbias[c];
    for (int i = 0; i < 256; ++i) a += pl[i] * cw[i*2 + c];
    out[3145728 + 32768 + m*2 + c] = a;
  }
}

// ---------------------------------------------------------------------------
// Host launcher
// ---------------------------------------------------------------------------

enum {
  I_SRC = 0, I_BOXES,
  I_T0_LNG, I_T0_LNB, I_T0_C1W, I_T0_C1B, I_T0_C2W, I_T0_C2B,
  I_T1_LNG, I_T1_LNB, I_T1_C1W, I_T1_C1B, I_T1_C2W, I_T1_C2B,
  I_QW, I_QB, I_KW, I_KB, I_VW, I_VB, I_OW, I_OB,
  I_FLNG, I_FLNB, I_FW1, I_FB1, I_FW2, I_FB2,
  I_SW1, I_SB1, I_SW2, I_SB2,
  I_PW1, I_PB1, I_PW2, I_PB2,
  I_MCW, I_MCB, I_MEW, I_MEB,
  I_GW, I_GB, I_LCG, I_LCB, I_CW, I_CB
};

extern "C" void kernel_launch(void* const* d_in, const int* in_sizes, int n_in,
                              void* d_out, int out_size, void* d_ws, size_t ws_size,
                              hipStream_t stream) {
  (void)in_sizes; (void)n_in; (void)out_size; (void)ws_size;
  char* base = (char*)d_ws;
  size_t off = 0;
  auto alloc = [&](size_t bytes) -> char* {
    char* p = base + off;
    off += (bytes + 255) & ~(size_t)255;
    return p;
  };

  float*    FT    = (float*)   alloc((size_t)1048576 * 4);
  float*    Y     = (float*)   alloc((size_t)3145728 * 4);
  _Float16* HLN   = (_Float16*)alloc((size_t)3145728 * 2);
  _Float16* H1    = (_Float16*)alloc((size_t)3145728 * 2);
  _Float16* YH    = (_Float16*)alloc((size_t)3145728 * 2);
  _Float16* SH    = (_Float16*)alloc((size_t)2097152 * 2);
  _Float16* QWs   = (_Float16*)alloc((size_t)1048576 * 2);
  _Float16* KWs   = (_Float16*)alloc((size_t)2097152 * 2);
  _Float16* VTs   = (_Float16*)alloc((size_t)2097152 * 2);
  _Float16* OWs   = (_Float16*)alloc((size_t)1048576 * 2);
  _Float16* HL2   = (_Float16*)alloc((size_t)1048576 * 2);
  _Float16* T1    = (_Float16*)alloc((size_t)4194304 * 2);
  _Float16* SIN   = (_Float16*)alloc((size_t)4194304 * 2);
  _Float16* S1    = (_Float16*)alloc((size_t)524288  * 2);
  _Float16* MODIN = (_Float16*)alloc((size_t)1835008 * 2);
  float*    CLN   = (float*)   alloc((size_t)1048576 * 4);
  float*    MODC  = (float*)   alloc((size_t)2097152 * 4);
  float*    MODE  = (float*)   alloc((size_t)2097152 * 4);
  float*    PE    = (float*)   alloc((size_t)1024    * 4);
  _Float16* WC1   = (_Float16*)alloc((size_t)3538944 * 2);  // 2 layers [768][2304]
  _Float16* WC2   = (_Float16*)alloc((size_t)1179648 * 2);  // 2 layers [768][768]
  _Float16* WQT   = (_Float16*)alloc((size_t)65536   * 2);
  _Float16* WKT   = (_Float16*)alloc((size_t)65536   * 2);
  _Float16* WVT   = (_Float16*)alloc((size_t)65536   * 2);
  _Float16* WOT   = (_Float16*)alloc((size_t)65536   * 2);
  _Float16* WF1T  = (_Float16*)alloc((size_t)262144  * 2);  // [1024][256]
  _Float16* WF2T  = (_Float16*)alloc((size_t)262144  * 2);  // [256][1024]
  _Float16* WS1T  = (_Float16*)alloc((size_t)131072  * 2);  // [128][1024]
  _Float16* WS2T  = (_Float16*)alloc((size_t)16384   * 2);  // [128][128]
  _Float16* WMCT  = (_Float16*)alloc((size_t)229376  * 2);  // [512][448]
  _Float16* WMET  = (_Float16*)alloc((size_t)229376  * 2);

  const float* src   = (const float*)d_in[I_SRC];
  const float* boxes = (const float*)d_in[I_BOXES];
  float* out = (float*)d_out;
  auto F = [&](int i) { return (const float*)d_in[i]; };

  // --- weight conversion ---
  k_feat_t<<<4096, 256, 0, stream>>>(src, FT);
  k_convC1<<<(768*2304 + 255)/256, 256, 0, stream>>>(WC1,              F(I_T0_C1W));
  k_convC1<<<(768*2304 + 255)/256, 256, 0, stream>>>(WC1 + 768*2304,   F(I_T1_C1W));
  k_convP <<<(768*768  + 255)/256, 256, 0, stream>>>(WC2,              F(I_T0_C2W), 768*768);
  k_convP <<<(768*768  + 255)/256, 256, 0, stream>>>(WC2 + 768*768,    F(I_T1_C2W), 768*768);
  k_convT <<<(65536  + 255)/256, 256, 0, stream>>>(WQT,  F(I_QW),  256, 256, 256);
  k_convT <<<(65536  + 255)/256, 256, 0, stream>>>(WKT,  F(I_KW),  256, 256, 256);
  k_convT <<<(65536  + 255)/256, 256, 0, stream>>>(WVT,  F(I_VW),  256, 256, 256);
  k_convT <<<(65536  + 255)/256, 256, 0, stream>>>(WOT,  F(I_OW),  256, 256, 256);
  k_convT <<<(262144 + 255)/256, 256, 0, stream>>>(WF1T, F(I_FW1), 1024, 256, 256);
  k_convT <<<(262144 + 255)/256, 256, 0, stream>>>(WF2T, F(I_FW2), 256, 1024, 1024);
  k_convT <<<(131072 + 255)/256, 256, 0, stream>>>(WS1T, F(I_SW1), 128, 1024, 1024);
  k_convT <<<(16384  + 255)/256, 256, 0, stream>>>(WS2T, F(I_SW2), 128, 128, 128);
  k_convT <<<(229376 + 255)/256, 256, 0, stream>>>(WMCT, F(I_MCW), 512, 416, 448);
  k_convT <<<(229376 + 255)/256, 256, 0, stream>>>(WMET, F(I_MEW), 512, 416, 448);
  k_pe<<<1, 256, 0, stream>>>(F(I_PW1), F(I_PB1), F(I_PW2), F(I_PB2), PE);

  // --- sample taps -> Y (Zs) ---
  k_sample<<<4096, 256, 0, stream>>>(FT, boxes, Y);

  // --- TCN layers ---
  for (int i = 0; i < 2; ++i) {
    const float* lng = F(i ? I_T1_LNG : I_T0_LNG);
    const float* lnb = F(i ? I_T1_LNB : I_T0_LNB);
    const float* c1b = F(i ? I_T1_C1B : I_T0_C1B);
    const float* c2b = F(i ? I_T1_C2B : I_T0_C2B);
    k_ln768<<<4096, 256, 0, stream>>>(Y, lng, lnb, HLN);
    k_tcn1<<<256, 256, 0, stream>>>(WC1 + (size_t)i*768*2304, c1b, HLN, H1, 1 << i);
    k_tcn2<<<256, 256, 0, stream>>>(WC2 + (size_t)i*768*768,  c2b, H1, Y);
  }

  // --- attention ---
  k_y2h<<<20480, 256, 0, stream>>>(Y, YH, SH);
  k_qkv<<<2560, 256, 0, stream>>>(YH, SH, WQT, F(I_QB), WKT, F(I_KB), WVT, F(I_VB),
                                  QWs, KWs, VTs);
  k_attn<<<1024, 64, 0, stream>>>(QWs, KWs, VTs, OWs);
  k_oproj<<<512, 256, 0, stream>>>(OWs, WOT, F(I_OB), Y);

  // --- FFN ---
  k_ln256<<<4096, 256, 0, stream>>>(Y, F(I_FLNG), F(I_FLNB), HL2, 256, nullptr);
  k_ffn1<<<2048, 256, 0, stream>>>(HL2, WF1T, F(I_FB1), T1);
  k_ffn2<<<512, 256, 0, stream>>>(T1, WF2T, F(I_FB2), Y);

  // --- SRM branch ---
  k_srm_in<<<4096, 256, 0, stream>>>(Y, SIN);
  k_srm1<<<256, 256, 0, stream>>>(SIN, WS1T, F(I_SB1), S1);
  k_srm2<<<256, 256, 0, stream>>>(S1, WS2T, F(I_SB2), MODIN);

  // --- modulation ---
  k_ln256<<<4096, 256, 0, stream>>>(Y, F(I_LCG), F(I_LCB), MODIN, 448, CLN);
  k_modin_pe<<<4096, 64, 0, stream>>>(PE, MODIN);
  k_mod<<<2048, 256, 0, stream>>>(MODIN, WMCT, F(I_MCB), WMET, F(I_MEB), MODC, MODE);
  k_combine<<<4096, 256, 0, stream>>>(MODIN, F(I_GW), F(I_GB), CLN, MODC, MODE, Y);

  // --- outputs ---
  k_out_copy<<<12288, 256, 0, stream>>>(Y, out);
  k_pool<<<128, 256, 0, stream>>>(Y, F(I_CW), F(I_CB), out);
}
